// Model_47545287967513
// MI455X (gfx1250) — matrix operand order repaired and checked by reading, not compiled, not run on hardware
//
#include <hip/hip_runtime.h>

// ---------------------------------------------------------------------------
// Problem constants
// ---------------------------------------------------------------------------
#define BB 64
#define SS 1024
#define DD 768
#define NA 24
#define NO 24
#define LL 8
#define NCAT 13
#define NPOL 3
#define RA (BB*NA)           // 1536 span rows per branch
#define RP (BB*NA*NO)        // 36864 pair rows
#define D2 (DD/2)            // 384

// Output offsets (flat concatenation in return order)
#define O_SEQ   0u
#define O_MATCH 327680u
#define O_ACAT  364544u
#define O_OCAT  384512u
#define O_XCAT  404480u
#define O_APOL  883712u
#define O_OPOL  888320u
#define O_XPOL  892928u

typedef __attribute__((ext_vector_type(16))) __bf16        v16bf;
typedef __attribute__((ext_vector_type(8)))  float         v8f;
typedef __attribute__((ext_vector_type(4)))  unsigned int  u32x4;
typedef __attribute__((ext_vector_type(8)))  int           i32x8;
typedef __attribute__((ext_vector_type(4)))  int           i32x4;
typedef unsigned short ushort_t;

union FragBf { v16bf v; u32x4 q[2]; };
union FragF  { v8f   v; float f[8]; };

__device__ __forceinline__ ushort_t f2bf(float x) {
    unsigned int u = __float_as_uint(x);
    unsigned int r = u + 0x7FFFu + ((u >> 16) & 1u);   // round-to-nearest-even
    return (ushort_t)(r >> 16);
}
__device__ __forceinline__ float bf2f(ushort_t h) {
    return __uint_as_float(((unsigned int)h) << 16);
}
__device__ __forceinline__ float selu_f(float x) {
    const float scale = 1.0507009873554805f, alpha = 1.6732632423543772f;
    return x > 0.f ? scale * x : scale * alpha * (__expf(x) - 1.f);
}
__device__ __forceinline__ float wave_sum(float v) {
    #pragma unroll
    for (int m = 16; m > 0; m >>= 1) v += __shfl_xor(v, m, 32);
    return v;
}

// ---------------------------------------------------------------------------
// K1: wq_eff[k] = sum_n Wk[k][n]*q[n] ; wq_eff[768] = bk.q + qb
// ---------------------------------------------------------------------------
__global__ void wq_eff_kernel(const float* __restrict__ Wk, const float* __restrict__ q,
                              const float* __restrict__ bk, const float* __restrict__ qb,
                              float* __restrict__ out) {
    int k = blockIdx.x * 256 + threadIdx.x;
    if (k < DD) {
        const float* row = Wk + (size_t)k * DD;
        float s = 0.f;
        #pragma unroll 4
        for (int n = 0; n < DD; ++n) s += row[n] * q[n];
        out[k] = s;
    }
    if (blockIdx.x == 0 && threadIdx.x == 0) {
        float c = 0.f;
        for (int n = 0; n < DD; ++n) c += bk[n] * q[n];
        out[DD] = c + qb[0];
    }
}

// ---------------------------------------------------------------------------
// K2: span collapse. One block per span row r = b*24+i.
// ---------------------------------------------------------------------------
__global__ void span_collapse_kernel(const float* __restrict__ emb,
                                     const int* __restrict__ idx,
                                     const float* __restrict__ wq,   // [769]
                                     ushort_t* __restrict__ ghat_bf, // [RA*DD]
                                     float* __restrict__ shat) {
    __shared__ float gbuf[LL][DD];
    __shared__ float att[LL];
    int r = blockIdx.x;
    int b = r / NA, i = r % NA;
    int wave = threadIdx.x >> 5, lane = threadIdx.x & 31;

    // stage A: each wave handles one span token l (float4 stream + dot)
    int t = idx[(b * NA + i) * LL + wave];
    const float4* row4 = (const float4*)(emb + ((size_t)b * SS + t) * DD);
    const float4* wq4 = (const float4*)wq;
    float4* gb4 = (float4*)(&gbuf[wave][0]);
    float acc = 0.f;
    #pragma unroll
    for (int s = 0; s < 6; ++s) {
        int i4 = s * 32 + lane;
        float4 g = row4[i4];
        float4 w = wq4[i4];
        gb4[i4] = g;
        acc += g.x * w.x + g.y * w.y + g.z * w.z + g.w * w.w;
    }
    acc = wave_sum(acc);
    if (lane == 0) att[wave] = acc + wq[DD];
    __syncthreads();

    // stage B: weighted sum over l
    for (int n = threadIdx.x; n < DD; n += 256) {
        float s = 0.f;
        #pragma unroll
        for (int l = 0; l < LL; ++l) s += att[l] * gbuf[l][n];
        ghat_bf[(size_t)r * DD + n] = f2bf(s);
    }
    if (threadIdx.x == 0) {
        float ss = 0.f;
        #pragma unroll
        for (int l = 0; l < LL; ++l) ss += att[l];
        shat[r] = ss;
    }
}

// ---------------------------------------------------------------------------
// K3: transpose+convert weights: Wt[n*K+k] = bf16(W[k*N+n])
// ---------------------------------------------------------------------------
__global__ void transpose_bf16_kernel(const float* __restrict__ W, ushort_t* __restrict__ Wt,
                                      int K, int N) {
    int id = blockIdx.x * 256 + threadIdx.x;
    if (id < N * K) {
        int n = id / K, k = id - n * K;
        Wt[id] = f2bf(W[(size_t)k * N + n]);
    }
}

// ---------------------------------------------------------------------------
// K4: generic WMMA bf16 GEMM.  C(M x Nstride) = A(M x 768) @ Bt(N x 768)^T
// ---------------------------------------------------------------------------
__global__ void gemm_bf16_kernel(const ushort_t* __restrict__ A,
                                 const ushort_t* __restrict__ Bt,
                                 const float* __restrict__ bias,
                                 const float* __restrict__ rowscale,
                                 float* __restrict__ Cf,
                                 ushort_t* __restrict__ Cbf,
                                 int Nstride) {
    const int K = DD;
    int mt   = blockIdx.x;
    int wave = threadIdx.x >> 5;
    int lane = threadIdx.x & 31;
    int nt   = blockIdx.y * 8 + wave;
    int mr   = lane & 15;
    int ksel = lane >> 4;

    const ushort_t* ap = A  + ((size_t)(mt * 16 + mr)) * K + ksel * 8;
    const ushort_t* bp = Bt + ((size_t)(nt * 16 + mr)) * K + ksel * 8;

    FragF acc;
    #pragma unroll
    for (int i = 0; i < 8; ++i) acc.f[i] = 0.f;

    #pragma unroll 4
    for (int kt = 0; kt < K / 32; ++kt) {
        int k0 = kt * 32;
        FragBf a, b;
        a.q[0] = *(const u32x4*)(ap + k0);
        a.q[1] = *(const u32x4*)(ap + k0 + 16);
        b.q[0] = *(const u32x4*)(bp + k0);
        b.q[1] = *(const u32x4*)(bp + k0 + 16);
        acc.v = __builtin_amdgcn_wmma_f32_16x16x32_bf16(false, a.v, false, b.v,
                                                        (short)0, acc.v, false, false);
    }

    int n = nt * 16 + mr;
    float badd = bias ? bias[n] : 0.f;
    #pragma unroll
    for (int v = 0; v < 8; ++v) {
        int m = mt * 16 + v + 8 * ksel;
        float add = bias ? (rowscale ? rowscale[m] * badd : badd) : 0.f;
        float val = acc.f[v] + add;
        Cf[(size_t)m * Nstride + n] = val;
        if (Cbf) Cbf[(size_t)m * Nstride + n] = f2bf(val);
    }
}

// ---------------------------------------------------------------------------
// K5: seq_score = emb @ seq_W + seq_b  (wave/row, float4 stream + prefetch)
// ---------------------------------------------------------------------------
__global__ void seq_kernel(const float* __restrict__ emb, const float* __restrict__ W,
                           const float* __restrict__ bvec, float* __restrict__ out) {
    int r = blockIdx.x * 8 + (threadIdx.x >> 5);
    int lane = threadIdx.x & 31;
    const float4* row4 = (const float4*)(emb + (size_t)r * DD);
    if (r + 8 < BB * SS)
        __builtin_prefetch(emb + (size_t)(r + 8) * DD + lane * 24, 0, 1);
    float acc[5] = {0.f, 0.f, 0.f, 0.f, 0.f};
    #pragma unroll
    for (int s = 0; s < 6; ++s) {
        float4 e = row4[s * 32 + lane];
        const float* w = W + (s * 32 + lane) * 4 * 5;
        #pragma unroll
        for (int c = 0; c < 5; ++c)
            acc[c] += e.x * w[c] + e.y * w[5 + c] + e.z * w[10 + c] + e.w * w[15 + c];
    }
    #pragma unroll
    for (int c = 0; c < 5; ++c) acc[c] = wave_sum(acc[c]);
    if (lane == 0) {
        #pragma unroll
        for (int c = 0; c < 5; ++c) out[(size_t)r * 5 + c] = acc[c] + bvec[c];
    }
}

// ---------------------------------------------------------------------------
// K6: cat/pol heads (wave per row)
// ---------------------------------------------------------------------------
__global__ void heads_kernel(const float* __restrict__ feat,
                             const float* __restrict__ catW, const float* __restrict__ catB,
                             const float* __restrict__ polW, const float* __restrict__ polB,
                             float* __restrict__ s_cat, float* __restrict__ s_pol) {
    int r = blockIdx.x * 8 + (threadIdx.x >> 5);
    int lane = threadIdx.x & 31;
    const float* row = feat + (size_t)r * DD;
    float acc[NCAT + NPOL];
    #pragma unroll
    for (int c = 0; c < NCAT + NPOL; ++c) acc[c] = 0.f;
    for (int k = lane; k < DD; k += 32) {
        float f = row[k];
        const float* cw = catW + k * NCAT;
        #pragma unroll
        for (int c = 0; c < NCAT; ++c) acc[c] += f * cw[c];
        const float* pw = polW + k * NPOL;
        #pragma unroll
        for (int c = 0; c < NPOL; ++c) acc[NCAT + c] += f * pw[c];
    }
    #pragma unroll
    for (int c = 0; c < NCAT + NPOL; ++c) acc[c] = wave_sum(acc[c]);
    if (lane == 0) {
        #pragma unroll
        for (int c = 0; c < NCAT; ++c) s_cat[(size_t)r * NCAT + c] = acc[c] + catB[c];
        #pragma unroll
        for (int c = 0; c < NPOL; ++c) s_pol[(size_t)r * NPOL + c] = acc[NCAT + c] + polB[c];
    }
}

// ---------------------------------------------------------------------------
// K7: cross_cat / cross_pol broadcast sums
// ---------------------------------------------------------------------------
__global__ void cross_kernel(const float* __restrict__ acat, const float* __restrict__ ocat,
                             const float* __restrict__ apol, const float* __restrict__ opol,
                             float* __restrict__ xcat, float* __restrict__ xpol) {
    int id = blockIdx.x * 256 + threadIdx.x;
    const int NXC = BB * NA * NO * NCAT;   // 479232
    if (id < NXC) {
        int c = id % NCAT;
        int j = (id / NCAT) % NO;
        int i = (id / (NCAT * NO)) % NA;
        int b = id / (NCAT * NO * NA);
        xcat[id] = acat[((size_t)b * NA + i) * NCAT + c] + ocat[((size_t)b * NO + j) * NCAT + c];
    } else {
        int id2 = id - NXC;                 // < 110592
        int c = id2 % NPOL;
        int j = (id2 / NPOL) % NO;
        int i = (id2 / (NPOL * NO)) % NA;
        int b = id2 / (NPOL * NO * NA);
        xpol[id2] = apol[((size_t)b * NA + i) * NPOL + c] + opol[((size_t)b * NO + j) * NPOL + c];
    }
}

// ---------------------------------------------------------------------------
// K8: fused pair MLP. One block per 64 pair rows (4 WMMA M-tiles).
//   h1 = selu(pa[rowA] + po[rowO])     (pa already contains m1_b)
//   h2 = selu(h1 @ m2 + m2_b)          WMMA, A from LDS, 4 acc per B-frag
//   match = h2 @ m3 + m3_b             (m3 staged in LDS via TDM)
// LDS (dynamic): h1 4x16x776 bf16 | h2 64x392 bf16 | m3 384 f32
// ---------------------------------------------------------------------------
#define H1S (DD + 8)                    // 776 (row = 388 DW -> bank offset 4)
#define H2S (D2 + 8)                    // 392
#define H1_USHORTS (4 * 16 * H1S)       // 49664
#define H2_USHORTS (64 * H2S)           // 25088
#define M3_OFF_BYTES (2 * (H1_USHORTS + H2_USHORTS))   // 149504
#define PAIR_SMEM_BYTES (M3_OFF_BYTES + D2 * 4)        // 151040
__global__ void pair_match_kernel(const float* __restrict__ pa, const float* __restrict__ po,
                                  const ushort_t* __restrict__ m2T, const float* __restrict__ m2b,
                                  const float* __restrict__ m3W, const float* __restrict__ m3b,
                                  float* __restrict__ match) {
    extern __shared__ char smem[];
    ushort_t* h1 = (ushort_t*)smem;
    ushort_t* h2 = h1 + H1_USHORTS;
    float*    m3s = (float*)(smem + M3_OFF_BYTES);

    int wave = threadIdx.x >> 5, lane = threadIdx.x & 31;
    int pr0 = blockIdx.x * 64;

    // --- stage m3_W (384 f32) into LDS via Tensor Data Mover ---
#if __has_builtin(__builtin_amdgcn_tensor_load_to_lds) && __has_builtin(__builtin_amdgcn_s_wait_tensorcnt)
    if (wave == 0) {
        // Tensor DMA descriptor (ISA 8.3/8.4): 1 valid desc, type=image(2),
        // data_size=4B, 1-row tile of 384 elements.
        unsigned long long ga = (unsigned long long)(const void*)m3W;
        u32x4 g0 = { 1u,                                    // count=1, user mode
                     (unsigned)M3_OFF_BYTES,                // lds_addr
                     (unsigned)(ga & 0xffffffffull),        // global_addr lo
                     (unsigned)((ga >> 32) & 0x01ffffffu) | (2u << 30) };  // hi | type=2
        i32x8 g1 = { (int)(2u << 16),        // data_size=4B
                     (int)((unsigned)D2 << 16),   // tensor_dim0 lo16 (=384)
                     (int)(1u << 16),        // tensor_dim0 hi=0 | tensor_dim1=1
                     (int)((unsigned)D2 << 16),   // tensor_dim1 hi=0 | tile_dim0=384
                     1,                      // tile_dim1=1, tile_dim2=0
                     D2,                     // tensor_dim0_stride=384
                     0, 0 };
        i32x4 gz4 = { 0, 0, 0, 0 };
        i32x8 gz8 = { 0, 0, 0, 0, 0, 0, 0, 0 };
        __builtin_amdgcn_tensor_load_to_lds(g0, g1, gz4, gz4, gz8, 0);
    }
#else
    for (int k = threadIdx.x; k < D2; k += 256) m3s[k] = m3W[k];
#endif

    // --- stage 1: h1 = selu(pa + po) for 64 pair rows, bf16 in LDS ---
    for (int e = threadIdx.x; e < 64 * DD; e += 256) {
        int r64 = e / DD, n = e - r64 * DD;
        int pr = pr0 + r64;
        int j = pr % NO;
        int bi = pr / NO;                 // = b*NA + i
        float x = pa[(size_t)bi * DD + n] + po[((size_t)(bi / NA) * NO + j) * DD + n];
        h1[(r64 >> 4) * (16 * H1S) + (r64 & 15) * H1S + n] = f2bf(selu_f(x));
    }
#if __has_builtin(__builtin_amdgcn_tensor_load_to_lds) && __has_builtin(__builtin_amdgcn_s_wait_tensorcnt)
    if (wave == 0) __builtin_amdgcn_s_wait_tensorcnt(0);
#endif
    __syncthreads();

    // --- stage 2: h2 = selu(h1 @ m2 + m2_b); 4 M-tiles per B-fragment ---
    int mr = lane & 15, ksel = lane >> 4;
    #pragma unroll
    for (int t = 0; t < 3; ++t) {
        int nt = wave * 3 + t;
        int col = nt * 16 + mr;
        const ushort_t* bp = m2T + (size_t)col * DD + ksel * 8;
        FragF acc[4];
        #pragma unroll
        for (int mt2 = 0; mt2 < 4; ++mt2)
            #pragma unroll
            for (int i = 0; i < 8; ++i) acc[mt2].f[i] = 0.f;
        for (int kt = 0; kt < DD / 32; ++kt) {
            int k0 = kt * 32;
            FragBf b;
            b.q[0] = *(const u32x4*)(bp + k0);
            b.q[1] = *(const u32x4*)(bp + k0 + 16);
            #pragma unroll
            for (int mt2 = 0; mt2 < 4; ++mt2) {
                const ushort_t* ap = h1 + mt2 * (16 * H1S) + mr * H1S + ksel * 8;
                FragBf a;
                a.q[0] = *(const u32x4*)(ap + k0);
                a.q[1] = *(const u32x4*)(ap + k0 + 16);
                acc[mt2].v = __builtin_amdgcn_wmma_f32_16x16x32_bf16(
                    false, a.v, false, b.v, (short)0, acc[mt2].v, false, false);
            }
        }
        float bb = m2b[col];
        #pragma unroll
        for (int mt2 = 0; mt2 < 4; ++mt2)
            #pragma unroll
            for (int v = 0; v < 8; ++v)
                h2[(mt2 * 16 + v + 8 * ksel) * H2S + col] = f2bf(selu_f(acc[mt2].f[v] + bb));
    }
    __syncthreads();

    // --- stage 3: match = h2 @ m3 + m3_b (8 waves x 8 rows each) ---
    float b3 = m3b[0];
    for (int m64 = wave; m64 < 64; m64 += 8) {
        float acc = 0.f;
        #pragma unroll
        for (int s = 0; s < D2 / 32; ++s) {
            int k = s * 32 + lane;
            acc += bf2f(h2[m64 * H2S + k]) * m3s[k];
        }
        acc = wave_sum(acc);
        if (lane == 0) match[pr0 + m64] = acc + b3;
    }
}

// ---------------------------------------------------------------------------
// Host launcher
// ---------------------------------------------------------------------------
extern "C" void kernel_launch(void* const* d_in, const int* in_sizes, int n_in,
                              void* d_out, int out_size, void* d_ws, size_t ws_size,
                              hipStream_t stream) {
    (void)in_sizes; (void)n_in; (void)out_size; (void)ws_size;
    const float* emb   = (const float*)d_in[0];
    const int*   aidx  = (const int*)d_in[1];
    const int*   oidx  = (const int*)d_in[2];
    const float* seqW  = (const float*)d_in[3];
    const float* seqb  = (const float*)d_in[4];
    const float* akW   = (const float*)d_in[5];
    const float* akb   = (const float*)d_in[6];
    const float* avW   = (const float*)d_in[7];
    const float* avb   = (const float*)d_in[8];
    const float* aqW   = (const float*)d_in[9];
    const float* aqb   = (const float*)d_in[10];
    const float* acatW = (const float*)d_in[11];
    const float* acatb = (const float*)d_in[12];
    const float* apolW = (const float*)d_in[13];
    const float* apolb = (const float*)d_in[14];
    const float* okW   = (const float*)d_in[15];
    const float* okb   = (const float*)d_in[16];
    const float* ovW   = (const float*)d_in[17];
    const float* ovb   = (const float*)d_in[18];
    const float* oqW   = (const float*)d_in[19];
    const float* oqb   = (const float*)d_in[20];
    const float* ocatW = (const float*)d_in[21];
    const float* ocatb = (const float*)d_in[22];
    const float* opolW = (const float*)d_in[23];
    const float* opolb = (const float*)d_in[24];
    const float* m1W   = (const float*)d_in[25];
    const float* m1b   = (const float*)d_in[26];
    const float* m2W   = (const float*)d_in[27];
    const float* m2b   = (const float*)d_in[28];
    const float* m3W   = (const float*)d_in[29];
    const float* m3b   = (const float*)d_in[30];
    float* out = (float*)d_out;

    // workspace carve-out (256B aligned)
    size_t off = 0;
    char* ws = (char*)d_ws;
    auto wsAlloc = [&](size_t bytes) -> void* {
        void* p = ws + off;
        off += (bytes + 255) & ~(size_t)255;
        return p;
    };
    float*    wqA    = (float*)wsAlloc(769 * sizeof(float));
    float*    wqO    = (float*)wsAlloc(769 * sizeof(float));
    float*    shatA  = (float*)wsAlloc(RA * sizeof(float));
    float*    shatO  = (float*)wsAlloc(RA * sizeof(float));
    ushort_t* ghatA  = (ushort_t*)wsAlloc((size_t)RA * DD * 2);
    ushort_t* ghatO  = (ushort_t*)wsAlloc((size_t)RA * DD * 2);
    ushort_t* WvaT   = (ushort_t*)wsAlloc((size_t)DD * DD * 2);
    ushort_t* WvoT   = (ushort_t*)wsAlloc((size_t)DD * DD * 2);
    ushort_t* m1aT   = (ushort_t*)wsAlloc((size_t)DD * DD * 2);
    ushort_t* m1bT   = (ushort_t*)wsAlloc((size_t)DD * DD * 2);
    ushort_t* m2T    = (ushort_t*)wsAlloc((size_t)D2 * DD * 2);
    float*    aout   = (float*)wsAlloc((size_t)RA * DD * 4);
    float*    oout   = (float*)wsAlloc((size_t)RA * DD * 4);
    ushort_t* aoutbf = (ushort_t*)wsAlloc((size_t)RA * DD * 2);
    ushort_t* ooutbf = (ushort_t*)wsAlloc((size_t)RA * DD * 2);
    float*    pa     = (float*)wsAlloc((size_t)RA * DD * 4);
    float*    po     = (float*)wsAlloc((size_t)RA * DD * 4);

    // allow >64KB dynamic LDS for the fused pair kernel
    (void)hipFuncSetAttribute((const void*)pair_match_kernel,
                              hipFuncAttributeMaxDynamicSharedMemorySize, PAIR_SMEM_BYTES);

    // K1: effective attention query vectors
    wq_eff_kernel<<<3, 256, 0, stream>>>(akW, aqW, akb, aqb, wqA);
    wq_eff_kernel<<<3, 256, 0, stream>>>(okW, oqW, okb, oqb, wqO);

    // K3: bf16 transposed weight copies (N-major => contiguous B-fragments)
    transpose_bf16_kernel<<<(DD * DD + 255) / 256, 256, 0, stream>>>(avW, WvaT, DD, DD);
    transpose_bf16_kernel<<<(DD * DD + 255) / 256, 256, 0, stream>>>(ovW, WvoT, DD, DD);
    transpose_bf16_kernel<<<(DD * DD + 255) / 256, 256, 0, stream>>>(m1W, m1aT, DD, DD);
    transpose_bf16_kernel<<<(DD * DD + 255) / 256, 256, 0, stream>>>(m1W + (size_t)DD * DD, m1bT, DD, DD);
    transpose_bf16_kernel<<<(DD * D2 + 255) / 256, 256, 0, stream>>>(m2W, m2T, DD, D2);

    // K5: seq_score (bandwidth-bound pass over bert_emb)
    seq_kernel<<<(BB * SS) / 8, 256, 0, stream>>>(emb, seqW, seqb, out + O_SEQ);

    // K2: span gather + attention collapse
    span_collapse_kernel<<<RA, 256, 0, stream>>>(emb, aidx, wqA, ghatA, shatA);
    span_collapse_kernel<<<RA, 256, 0, stream>>>(emb, oidx, wqO, ghatO, shatO);

    // K4: a_out = ghat @ Wv + shat*bv  (f32 + bf16 copies)
    dim3 ggrid(RA / 16, DD / 128);
    gemm_bf16_kernel<<<ggrid, 256, 0, stream>>>(ghatA, WvaT, avb, shatA, aout, aoutbf, DD);
    gemm_bf16_kernel<<<ggrid, 256, 0, stream>>>(ghatO, WvoT, ovb, shatO, oout, ooutbf, DD);

    // K6: category / polarity heads
    heads_kernel<<<RA / 8, 256, 0, stream>>>(aout, acatW, acatb, apolW, apolb,
                                             out + O_ACAT, out + O_APOL);
    heads_kernel<<<RA / 8, 256, 0, stream>>>(oout, ocatW, ocatb, opolW, opolb,
                                             out + O_OCAT, out + O_OPOL);

    // K7: cross sums (479232 + 110592 = 589824 elements = 2304 blocks)
    cross_kernel<<<2304, 256, 0, stream>>>(out + O_ACAT, out + O_OCAT,
                                           out + O_APOL, out + O_OPOL,
                                           out + O_XCAT, out + O_XPOL);

    // K4: pa = a_out @ m1W[:D] + m1_b ; po = o_out @ m1W[D:]
    gemm_bf16_kernel<<<ggrid, 256, 0, stream>>>(aoutbf, m1aT, m1b, nullptr, pa, nullptr, DD);
    gemm_bf16_kernel<<<ggrid, 256, 0, stream>>>(ooutbf, m1bT, nullptr, nullptr, po, nullptr, DD);

    // K8: fused pair MLP -> match_score
    pair_match_kernel<<<RP / 64, 256, PAIR_SMEM_BYTES, stream>>>(pa, po, m2T, m2b, m3W, m3b,
                                                                 out + O_MATCH);
}
